// InvariantPointAttention_77034533421608
// MI455X (gfx1250) — matrix operand order repaired and checked by reading, not compiled, not run on hardware
//
#include <hip/hip_runtime.h>
#include <hip/hip_bf16.h>
#include <math.h>

#define DI __device__ __forceinline__

typedef __attribute__((ext_vector_type(16))) _Float16 v16h;
typedef __attribute__((ext_vector_type(8)))  float    v8f;

constexpr int NN   = 512;
constexpr int C2D_ = 128;
constexpr int HH   = 12;
constexpr int KOUT = 1152;   // 192 + 384 + 144 + 432 projection cols
constexpr float SCALAR_Wc = 2.309401076758503f;    // sqrt(16/3)
constexpr float POINT_Wc  = 2.449489742783178f;    // sqrt(4*9/2/3) = sqrt(6)
constexpr float ATT2D_Wc  = 0.57735026918962576f;  // sqrt(1/3)
constexpr float EPSc = 1e-8f;
constexpr int OUTW = 2112;   // per-row output width

// ---------------- workspace layout (bytes) ----------------
constexpr size_t OFF_AFRAG = 0;                                   // 512x512 f16 (frag packed)
constexpr size_t OFF_WFRAG = OFF_AFRAG + (size_t)512*512*2;       // 1152x512 f16 (frag packed)
constexpr size_t OFF_CPROJ = OFF_WFRAG + (size_t)KOUT*512*2;      // 512x1152 f32
constexpr size_t OFF_WATT  = OFF_CPROJ + (size_t)512*KOUT*4;      // 16x128 f16
constexpr size_t OFF_ATT2D = OFF_WATT  + (size_t)16*128*2;        // 512*512 x 16 f32 (prescaled)
constexpr size_t OFF_QH    = OFF_ATT2D + (size_t)512*512*16*4;    // 12x512x32 f16
constexpr size_t OFF_KH    = OFF_QH    + (size_t)12*512*32*2;     // 12x512x32 f16
constexpr size_t OFF_BQ    = OFF_KH    + (size_t)12*512*32*2;     // 12x512 f32
constexpr size_t OFF_BK    = OFF_BQ    + (size_t)12*512*4;        // 12x512 f32
constexpr size_t OFF_VT    = OFF_BK    + (size_t)12*512*4;        // 12x48x512 f16
constexpr size_t OFF_PQ    = OFF_VT    + (size_t)12*48*512*2;     // 512x16x512 f16
constexpr size_t OFF_OVS   = OFF_PQ    + (size_t)512*16*512*2;    // 12x512x48 f32

// ---------------- CDNA5 async copy helpers ----------------
DI void async_wait0() {
#if __has_builtin(__builtin_amdgcn_s_wait_asynccnt)
  __builtin_amdgcn_s_wait_asynccnt(0);
#else
  asm volatile("s_wait_asynccnt 0x0" ::: "memory");
#endif
}

// per-lane 16B global -> LDS async copy (GVS mode: SGPR base + per-lane VGPR offset)
DI void async_copy16(unsigned lds_byte_addr, unsigned gbyte_off, const void* sbase) {
  asm volatile("global_load_async_to_lds_b128 %0, %1, %2"
               :: "v"(lds_byte_addr), "v"(gbyte_off), "s"(sbase) : "memory");
}

// ---------------- WMMA helpers (CDNA5 16x16x32 f16) ----------------
DI v8f wmma16(v16h a, v16h b, v8f c) {
  return __builtin_amdgcn_wmma_f32_16x16x32_f16(false, a, false, b, (short)0, c,
                                                false, false);
}

// A fragment (16x32 f16): row m = lane&15, lane half hl holds K = hl*16 .. hl*16+15.
DI v16h load_a(const _Float16* base, int row_stride, int k0, int lane) {
  int m = lane & 15, hl = lane >> 4;
  const _Float16* p = base + (size_t)m * row_stride + k0;
  v16h a;
#pragma unroll
  for (int v = 0; v < 8; ++v) {
    int kk = hl * 16 + v * 2;
    a[2*v]   = p[kk];
    a[2*v+1] = p[kk+1];
  }
  return a;
}

// B fragment (32x16): source stored "transposed" as [n][k] rows contiguous in k.
// lanes 0-15 hold K=0..15, lanes 16-31 hold K=16..31 -> one contiguous 32B load.
DI v16h load_bt(const _Float16* base, int row_stride, int k0, int lane) {
  int n = lane & 15, hl = lane >> 4;
  const _Float16* p = base + (size_t)n * row_stride + k0 + hl * 16;
  return *(const v16h*)p;
}

// ---------------- kernels ----------------

// Pack inputs_1d (f32 512x512) into A-fragment layout (f16).
__global__ void k_pack_a(const float* __restrict__ in1d, _Float16* __restrict__ afrag) {
  int mb = blockIdx.x, kb = blockIdx.y, lane = threadIdx.x;
  int m = lane & 15, hl = lane >> 4;
  const float* p = in1d + (size_t)(mb*16 + m) * NN + kb*32;
  v16h a;
#pragma unroll
  for (int v = 0; v < 8; ++v) {
    int kk = hl * 16 + v * 2;
    a[2*v]   = (_Float16)p[kk];
    a[2*v+1] = (_Float16)p[kk+1];
  }
  *(v16h*)(afrag + (((size_t)(mb*16 + kb))*32 + lane)*16) = a;
}

// Pack the four projection weight matrices (row-major [out,512]) into B-fragment layout.
__global__ void k_pack_w(const float* __restrict__ wqs, const float* __restrict__ wkvs,
                         const float* __restrict__ wqp, const float* __restrict__ wkvp,
                         _Float16* __restrict__ wfrag) {
  int nb = blockIdx.x, kb = blockIdx.y, lane = threadIdx.x;
  int n = lane & 15, hl = lane >> 4;
  int c = nb*16 + n;
  const float* w; int row;
  if (c < 192)      { w = wqs;  row = c; }
  else if (c < 576) { w = wkvs; row = c - 192; }
  else if (c < 720) { w = wqp;  row = c - 576; }
  else              { w = wkvp; row = c - 720; }
  const float* p = w + (size_t)row * NN + kb*32 + hl*16;
  v16h b;
#pragma unroll
  for (int j = 0; j < 16; ++j) b[j] = (_Float16)p[j];
  *(v16h*)(wfrag + (((size_t)(nb*16 + kb))*32 + lane)*16) = b;
}

// w_att2d (12x128) -> f16 [16][128], rows 12..15 zero.
__global__ void k_pack_watt(const float* __restrict__ watt, _Float16* __restrict__ wh) {
  int i = blockIdx.x * blockDim.x + threadIdx.x;   // 0..2047
  if (i >= 16*C2D_) return;
  int r = i >> 7, c = i & 127;
  wh[i] = (r < HH) ? (_Float16)watt[r*C2D_ + c] : (_Float16)0.f;
}

// zero a buffer in 16B chunks (for Pq padding rows h=12..15)
__global__ void k_zero(uint4* __restrict__ p, long n16) {
  long i = (long)blockIdx.x * blockDim.x + threadIdx.x;
  if (i < n16) p[i] = make_uint4(0u, 0u, 0u, 0u);
}

// Projection GEMM: Cproj[512][1152] = in1d * W^T + b   (WMMA, frag-packed operands)
__global__ void __launch_bounds__(32) k_proj(const _Float16* __restrict__ afrag,
                                             const _Float16* __restrict__ wfrag,
                                             const float* __restrict__ bqs, const float* __restrict__ bkvs,
                                             const float* __restrict__ bqp, const float* __restrict__ bkvp,
                                             float* __restrict__ cproj) {
  int mb = blockIdx.x, nb = blockIdx.y, lane = threadIdx.x;
  v8f acc = {};
  for (int kb = 0; kb < 16; ++kb) {
    v16h a = *(const v16h*)(afrag + (((size_t)(mb*16 + kb))*32 + lane)*16);
    v16h b = *(const v16h*)(wfrag + (((size_t)(nb*16 + kb))*32 + lane)*16);
    acc = wmma16(a, b, acc);
  }
  int n = lane & 15, hl = lane >> 4;
  int c = nb*16 + n;
  float bias;
  if (c < 192)      bias = bqs[c];
  else if (c < 576) bias = bkvs[c-192];
  else if (c < 720) bias = bqp[c-576];
  else              bias = bkvp[c-720];
#pragma unroll
  for (int r = 0; r < 8; ++r)
    cproj[(size_t)(mb*16 + r + hl*8) * KOUT + c] = acc[r] + bias;
}

// att2d: for every (q,k) row of inputs_2d, 128-dot with each head filter.
// abuf[(q*512+k)*16 + h] = ATT2D_W * (in2d[q,k,:] . watt[h,:] + b[h])
__global__ void __launch_bounds__(32) k_att2d(const float* __restrict__ in2d,
                                              const _Float16* __restrict__ wh,
                                              const float* __restrict__ batt,
                                              float* __restrict__ abuf) {
  int t = blockIdx.x;                 // qk row tile (16 rows), 16384 tiles
  int lane = threadIdx.x;
  int m = lane & 15, hl = lane >> 4;
  const float* arow = in2d + (size_t)(t*16 + m) * C2D_;
  v8f acc = {};
  for (int kb = 0; kb < 4; ++kb) {
    v16h a;
#pragma unroll
    for (int v = 0; v < 8; ++v) {
      int kk = kb*32 + hl*16 + v*2;
      a[2*v]   = (_Float16)arow[kk];
      a[2*v+1] = (_Float16)arow[kk+1];
    }
    v16h b = load_bt(wh, C2D_, kb*32, lane);
    acc = wmma16(a, b, acc);
  }
  int n = lane & 15;
  float bn = (n < HH) ? batt[n] : 0.f;
#pragma unroll
  for (int r = 0; r < 8; ++r)
    abuf[(size_t)(t*16 + r + hl*8) * 16 + n] = ATT2D_Wc * (acc[r] + bn);
}

// Build fused Q/K features (scalar ‖ w*point, pad 32), V^T (scalar ‖ point, pad 48),
// and the -0.5*w*|point|^2 bias vectors. One thread per (n,h).
__global__ void k_qkv(const float* __restrict__ cproj, const float* __restrict__ rot,
                      const float* __restrict__ trn, const float* __restrict__ tpw,
                      _Float16* __restrict__ Qh, _Float16* __restrict__ Kh,
                      float* __restrict__ bq, float* __restrict__ bk,
                      _Float16* __restrict__ VT) {
  int idx = blockIdx.x * blockDim.x + threadIdx.x;
  if (idx >= NN*HH) return;
  int n = idx / HH, h = idx % HH;
  const float* cp = cproj + (size_t)n * KOUT;
  float R[9], T[3];
#pragma unroll
  for (int i = 0; i < 9; ++i) R[i] = rot[n*9 + i];
#pragma unroll
  for (int i = 0; i < 3; ++i) T[i] = trn[n*3 + i];
  float w = tpw[h] * POINT_Wc;
  _Float16* qrow = Qh + ((size_t)h*NN + n) * 32;
  _Float16* krow = Kh + ((size_t)h*NN + n) * 32;
#pragma unroll
  for (int c = 0; c < 16; ++c) {
    qrow[c] = (_Float16)(SCALAR_Wc * cp[h*16 + c]);
    krow[c] = (_Float16)(cp[192 + h*32 + c]);
    VT[((size_t)h*48 + c)*NN + n] = (_Float16)(cp[192 + h*32 + 16 + c]);
  }
  float sq = 0.f;
#pragma unroll
  for (int p = 0; p < 4; ++p) {
    float l0 = cp[576 + 0*48 + h*4 + p];
    float l1 = cp[576 + 1*48 + h*4 + p];
    float l2 = cp[576 + 2*48 + h*4 + p];
#pragma unroll
    for (int i = 0; i < 3; ++i) {
      float g = R[i*3+0]*l0 + R[i*3+1]*l1 + R[i*3+2]*l2 + T[i];
      qrow[16 + p*3 + i] = (_Float16)(w * g);
      sq += g * g;
    }
  }
  bq[h*NN + n] = -0.5f * w * sq;
  float sk = 0.f;
#pragma unroll
  for (int d = 0; d < 12; ++d) {
    float l0 = cp[720 + 0*144 + h*12 + d];
    float l1 = cp[720 + 1*144 + h*12 + d];
    float l2 = cp[720 + 2*144 + h*12 + d];
#pragma unroll
    for (int i = 0; i < 3; ++i) {
      float g = R[i*3+0]*l0 + R[i*3+1]*l1 + R[i*3+2]*l2 + T[i];
      if (d < 4) { krow[16 + d*3 + i] = (_Float16)g; sk += g * g; }
      else       { VT[((size_t)h*48 + 16 + (d-4)*3 + i)*NN + n] = (_Float16)g; }
    }
  }
  bk[h*NN + n] = -0.5f * w * sk;
#pragma unroll
  for (int z = 28; z < 32; ++z) { qrow[z] = (_Float16)0.f; krow[z] = (_Float16)0.f; }
#pragma unroll
  for (int z = 40; z < 48; ++z) VT[((size_t)h*48 + z)*NN + n] = (_Float16)0.f;
}

// Attention logits + softmax. One wave per (head, 16-query tile).
// One 16x16x32 WMMA per key tile (fused scalar+point contraction), LDS staging,
// lane-shuffle row reductions, exp via v_exp_f32. Writes P f16 in [q][16][512].
__global__ void __launch_bounds__(32) k_attn(const _Float16* __restrict__ Qh,
                                             const _Float16* __restrict__ Kh,
                                             const float* __restrict__ bq, const float* __restrict__ bk,
                                             const float* __restrict__ abuf,
                                             const float* __restrict__ mask,
                                             _Float16* __restrict__ Pq) {
  __shared__ float lt[16 * NN];
  int h = blockIdx.x, mb = blockIdx.y, lane = threadIdx.x;
  int n = lane & 15, hl = lane >> 4;
  v16h a = load_a(Qh + ((size_t)h*NN + mb*16) * 32, 32, 0, lane);
  float rmax[8], mq[8], bqr[8];
#pragma unroll
  for (int r = 0; r < 8; ++r) {
    int q = mb*16 + r + hl*8;
    rmax[r] = -3.0e38f;
    mq[r]   = mask[q];
    bqr[r]  = bq[h*NN + q];
  }
  for (int kb = 0; kb < 32; ++kb) {
    v16h b = load_bt(Kh + ((size_t)h*NN + kb*16) * 32, 32, 0, lane);
    v8f acc = {};
    acc = wmma16(a, b, acc);
    int key = kb*16 + n;
    float bkk = bk[h*NN + key];
    float mk  = mask[key];
#pragma unroll
    for (int r = 0; r < 8; ++r) {
      int row = r + hl*8;
      int q = mb*16 + row;
      float x = acc[r] + bqr[r] + bkk
              + abuf[((size_t)q*NN + key)*16 + h]
              - 100000.f * (1.f - mq[r]*mk);
      lt[row*NN + key] = x;
      rmax[r] = fmaxf(rmax[r], x);
    }
  }
#pragma unroll
  for (int r = 0; r < 8; ++r)
#pragma unroll
    for (int off = 1; off < 16; off <<= 1)
      rmax[r] = fmaxf(rmax[r], __shfl_xor(rmax[r], off, 32));
  float rsum[8] = {0.f,0.f,0.f,0.f,0.f,0.f,0.f,0.f};
  for (int kb = 0; kb < 32; ++kb) {
    int key = kb*16 + n;
#pragma unroll
    for (int r = 0; r < 8; ++r) {
      int row = r + hl*8;
      float e = __expf(lt[row*NN + key] - rmax[r]);
      lt[row*NN + key] = e;
      rsum[r] += e;
    }
  }
#pragma unroll
  for (int r = 0; r < 8; ++r)
#pragma unroll
    for (int off = 1; off < 16; off <<= 1)
      rsum[r] += __shfl_xor(rsum[r], off, 32);
  float rinv[8];
#pragma unroll
  for (int r = 0; r < 8; ++r) rinv[r] = 1.f / rsum[r];
  for (int kb = 0; kb < 32; ++kb) {
    int key = kb*16 + n;
#pragma unroll
    for (int r = 0; r < 8; ++r) {
      int row = r + hl*8;
      int q = mb*16 + row;
      Pq[((size_t)q*16 + h)*NN + key] = (_Float16)(lt[row*NN + key] * rinv[r]);
    }
  }
}

// attn @ [v_scalar | v_point]: per (head, q-tile, 16-chan block of 48). WMMA over K=512.
__global__ void __launch_bounds__(32) k_attnv(const _Float16* __restrict__ Pq,
                                              const _Float16* __restrict__ VT,
                                              float* __restrict__ Ovs) {
  int h = blockIdx.x, mb = blockIdx.y, nb = blockIdx.z, lane = threadIdx.x;
  v8f acc = {};
  for (int kb = 0; kb < 16; ++kb) {
    v16h a = load_a(Pq + ((size_t)(mb*16)*16 + h)*NN, 16*NN, kb*32, lane);
    v16h b = load_bt(VT + ((size_t)h*48 + nb*16)*NN, NN, kb*32, lane);
    acc = wmma16(a, b, acc);
  }
  int n = lane & 15, hl = lane >> 4;
#pragma unroll
  for (int r = 0; r < 8; ++r)
    Ovs[((size_t)h*NN + mb*16 + r + hl*8)*48 + nb*16 + n] = acc[r];
}

// r2d: per query q, attnT_q (16x512, rows=heads padded) @ inputs_2d[q] (512x128).
// 8 waves per block. Double-buffered ASYNC global->LDS staging of 16KB f32 tiles
// (global_load_async_to_lds_b128 + s_wait_asynccnt), WMMA consumes tile kb while
// the async engine copies tile kb+1.
__global__ void __launch_bounds__(256) k_r2d(const _Float16* __restrict__ Pq,
                                             const float* __restrict__ in2d,
                                             float* __restrict__ out) {
  __shared__ float tile[2][32 * C2D_];   // [buf][k][c] f32, 2 x 16KB
  int q = blockIdx.x;
  int tid = threadIdx.x;
  int lane = tid & 31, wv = tid >> 5;    // wv = output channel block (0..7)
  int n = lane & 15, hl = lane >> 4;
  const float* srcq = in2d + (size_t)q * NN * C2D_;
  unsigned lbase = (unsigned)(size_t)(&tile[0][0]);
  v8f acc = {};

  // prologue: start async copy of tile 0 into buffer 0
#pragma unroll
  for (int it = 0; it < 4; ++it)
    async_copy16(lbase + (unsigned)(tid*16 + it*4096),
                 (unsigned)(tid*16 + it*4096), srcq);

  for (int kb = 0; kb < 16; ++kb) {
    int buf = kb & 1;
    async_wait0();        // this wave's copies for tile kb have landed
    __syncthreads();      // all waves' copies for tile kb have landed
    // kick off tile kb+1 into the other buffer (its last readers finished
    // at iteration kb-1, before the barrier above)
    if (kb + 1 < 16) {
      unsigned lb = lbase + (unsigned)((buf ^ 1) * 32 * C2D_ * 4);
      unsigned gb = (unsigned)((kb + 1) * 32 * C2D_ * 4);
#pragma unroll
      for (int it = 0; it < 4; ++it)
        async_copy16(lb + (unsigned)(tid*16 + it*4096),
                     gb + (unsigned)(tid*16 + it*4096), srcq);
    }
    // consume tile kb
    v16h a = load_a(Pq + (size_t)q*16*NN, NN, kb*32, lane);
    v16h b;
#pragma unroll
    for (int j = 0; j < 16; ++j)
      b[j] = (_Float16)tile[buf][(hl*16 + j)*C2D_ + wv*16 + n];
    acc = wmma16(a, b, acc);
  }
#pragma unroll
  for (int r = 0; r < 8; ++r) {
    int hrow = r + hl*8;
    if (hrow < HH)
      out[(size_t)q*OUTW + 576 + hrow*C2D_ + wv*16 + n] = acc[r];
  }
}

// Finalize: scalar output copy, frame-inverse point transform, dist.
__global__ void k_final(const float* __restrict__ Ovs, const float* __restrict__ rot,
                        const float* __restrict__ trn, float* __restrict__ out) {
  int idx = blockIdx.x * blockDim.x + threadIdx.x;
  if (idx >= NN*HH*16) return;
  int q = idx / (HH*16);
  int rem = idx % (HH*16);
  int h = rem / 16, c = rem % 16;
  const float* ov = Ovs + ((size_t)h*NN + q) * 48;
  out[(size_t)q*OUTW + h*16 + c] = ov[c];
  if (c < 8) {
    int p = c;
    float g0 = ov[16 + p*3 + 0] - trn[q*3 + 0];
    float g1 = ov[16 + p*3 + 1] - trn[q*3 + 1];
    float g2 = ov[16 + p*3 + 2] - trn[q*3 + 2];
    float l0 = rot[q*9 + 0]*g0 + rot[q*9 + 3]*g1 + rot[q*9 + 6]*g2;
    float l1 = rot[q*9 + 1]*g0 + rot[q*9 + 4]*g1 + rot[q*9 + 7]*g2;
    float l2 = rot[q*9 + 2]*g0 + rot[q*9 + 5]*g1 + rot[q*9 + 8]*g2;
    out[(size_t)q*OUTW + 192 + h*8 + p] = l0;
    out[(size_t)q*OUTW + 288 + h*8 + p] = l1;
    out[(size_t)q*OUTW + 384 + h*8 + p] = l2;
    out[(size_t)q*OUTW + 480 + h*8 + p] = sqrtf(EPSc + l0*l0 + l1*l1 + l2*l2);
  }
}

extern "C" void kernel_launch(void* const* d_in, const int* in_sizes, int n_in,
                              void* d_out, int out_size, void* d_ws, size_t ws_size,
                              hipStream_t stream) {
  (void)in_sizes; (void)n_in; (void)out_size; (void)ws_size;
  const float* in1d = (const float*)d_in[0];
  const float* in2d = (const float*)d_in[1];
  const float* mask = (const float*)d_in[2];
  const float* rot  = (const float*)d_in[3];
  const float* trn  = (const float*)d_in[4];
  const float* wqs  = (const float*)d_in[5];
  const float* bqs  = (const float*)d_in[6];
  const float* wkvs = (const float*)d_in[7];
  const float* bkvs = (const float*)d_in[8];
  const float* wqp  = (const float*)d_in[9];
  const float* bqp  = (const float*)d_in[10];
  const float* wkvp = (const float*)d_in[11];
  const float* bkvp = (const float*)d_in[12];
  const float* watt = (const float*)d_in[13];
  const float* batt = (const float*)d_in[14];
  const float* tpw  = (const float*)d_in[15];

  char* ws = (char*)d_ws;
  _Float16* afrag = (_Float16*)(ws + OFF_AFRAG);
  _Float16* wfrag = (_Float16*)(ws + OFF_WFRAG);
  float*    cproj = (float*)   (ws + OFF_CPROJ);
  _Float16* wh    = (_Float16*)(ws + OFF_WATT);
  float*    abuf  = (float*)   (ws + OFF_ATT2D);
  _Float16* Qh    = (_Float16*)(ws + OFF_QH);
  _Float16* Kh    = (_Float16*)(ws + OFF_KH);
  float*    bqv   = (float*)   (ws + OFF_BQ);
  float*    bkv   = (float*)   (ws + OFF_BK);
  _Float16* VT    = (_Float16*)(ws + OFF_VT);
  _Float16* pq    = (_Float16*)(ws + OFF_PQ);
  float*    ovs   = (float*)   (ws + OFF_OVS);
  float*    out   = (float*)d_out;

  k_pack_a   <<<dim3(32,16), 32, 0, stream>>>(in1d, afrag);
  k_pack_w   <<<dim3(72,16), 32, 0, stream>>>(wqs, wkvs, wqp, wkvp, wfrag);
  k_pack_watt<<<8, 256, 0, stream>>>(watt, wh);
  k_zero     <<<2048, 256, 0, stream>>>((uint4*)pq, (long)(512*16*512*2/16));
  k_proj     <<<dim3(32,72), 32, 0, stream>>>(afrag, wfrag, bqs, bkvs, bqp, bkvp, cproj);
  k_att2d    <<<16384, 32, 0, stream>>>(in2d, wh, batt, abuf);
  k_qkv      <<<(NN*HH + 255)/256, 256, 0, stream>>>(cproj, rot, trn, tpw, Qh, Kh, bqv, bkv, VT);
  k_attn     <<<dim3(HH,32), 32, 0, stream>>>(Qh, Kh, bqv, bkv, abuf, mask, pq);
  k_attnv    <<<dim3(HH,32,3), 32, 0, stream>>>(pq, VT, ovs);
  k_r2d      <<<512, 256, 0, stream>>>(pq, in2d, out);
  k_final    <<<(NN*HH*16 + 255)/256, 256, 0, stream>>>(ovs, rot, trn, out);
}
